// SGC_LL_89653147337354
// MI455X (gfx1250) — matrix operand order repaired and checked by reading, not compiled, not run on hardware
//
#include <hip/hip_runtime.h>
#include <hip/hip_bf16.h>

#define B_ 128
#define N_ 512
#define F_ 128
#define OUT_ 128
#define TINY_ 1.401298464324817e-45f

typedef __attribute__((ext_vector_type(16))) __bf16          v16bf;
typedef __attribute__((ext_vector_type(8)))  float           v8f;
typedef __attribute__((ext_vector_type(4)))  float           v4f;
typedef __attribute__((ext_vector_type(2)))  float           v2f;
typedef __attribute__((ext_vector_type(8)))  unsigned short  v8us;
typedef __attribute__((ext_vector_type(4)))  unsigned short  v4us;
typedef __attribute__((ext_vector_type(4)))  unsigned int    v4u;
typedef __attribute__((ext_vector_type(8)))  int             v8i;
typedef __attribute__((ext_vector_type(4)))  int             v4i;

// ---------- helpers ----------
__device__ __forceinline__ unsigned short bf2us(__bf16 h) {
  return __builtin_bit_cast(unsigned short, h);
}
__device__ __forceinline__ v8f vzero8() {
  v8f z;
#pragma unroll
  for (int i = 0; i < 8; ++i) z[i] = 0.f;
  return z;
}

union frag_u {
  v8us  u[2];
  unsigned short s[16];
  v16bf b;
};

// A fragment (16x32 bf16) from bf16 row-major: two contiguous 16B runs per lane
__device__ __forceinline__ v16bf a_frag_bf(const unsigned short* base, int ld, int k0) {
  int l = threadIdx.x & 31, m = l & 15, half = l >> 4;
  const unsigned short* rp = base + (size_t)m * ld + k0 + half * 8;
  frag_u t;
  t.u[0] = *(const v8us*)(rp);        // k = half*8 + 0..7
  t.u[1] = *(const v8us*)(rp + 16);   // k = half*8 + 16..23
  return t.b;
}

// A fragment from (L + laplacian) f32, fused add + native bf16 cvt
__device__ __forceinline__ v16bf a_frag_sum(const float* p, const float* q, int ld, int k0) {
  int l = threadIdx.x & 31, m = l & 15, half = l >> 4;
  size_t ro = (size_t)m * ld + k0 + half * 8;
  v4f s0 = *(const v4f*)(p + ro)      + *(const v4f*)(q + ro);
  v4f s1 = *(const v4f*)(p + ro + 4)  + *(const v4f*)(q + ro + 4);
  v4f s2 = *(const v4f*)(p + ro + 16) + *(const v4f*)(q + ro + 16);
  v4f s3 = *(const v4f*)(p + ro + 20) + *(const v4f*)(q + ro + 20);
  v16bf a;
#pragma unroll
  for (int i = 0; i < 4; ++i) {
    a[i]      = (__bf16)s0[i];
    a[4 + i]  = (__bf16)s1[i];
    a[8 + i]  = (__bf16)s2[i];
    a[12 + i] = (__bf16)s3[i];
  }
  return a;
}

// A fragment for xk (feature index = f*2 + k): interleave x / x1 rows
__device__ __forceinline__ v16bf a_frag_inter(const unsigned short* x0,
                                              const unsigned short* x1, int k0) {
  int l = threadIdx.x & 31, m = l & 15, half = l >> 4;
  int f0 = (k0 + half * 8) >> 1;
  const unsigned short* r0 = x0 + (size_t)m * F_ + f0;
  const unsigned short* r1 = x1 + (size_t)m * F_ + f0;
  v4us a0 = *(const v4us*)(r0),     b0 = *(const v4us*)(r1);
  v4us a1 = *(const v4us*)(r0 + 8), b1 = *(const v4us*)(r1 + 8);
  frag_u t;
#pragma unroll
  for (int i = 0; i < 8; ++i) {
    t.s[i]     = (i & 1) ? b0[i >> 1] : a0[i >> 1];
    t.s[8 + i] = (i & 1) ? b1[i >> 1] : a1[i >> 1];
  }
  return t.b;
}

// B fragment (32x16) from bf16 K-major (transposed) operand: contiguous in K
__device__ __forceinline__ v16bf b_frag_tr(const unsigned short* baseT, int ldT,
                                           int k0, int col0) {
  int l = threadIdx.x & 31, n = l & 15, half = l >> 4;
  const unsigned short* cp = baseT + (size_t)(col0 + n) * ldT + k0 + half * 8;
  frag_u t;
  t.u[0] = *(const v8us*)(cp);
  t.u[1] = *(const v8us*)(cp + 16);
  return t.b;
}

// ---------- kernels ----------
__global__ void k_zero(float* norm) {
  if (threadIdx.x < B_) norm[threadIdx.x] = 0.f;
}

// x -> bf16 row-major AND bf16 K-major (xT[b][f][n]) via LDS tile transpose
__global__ void k_cvt(const float* __restrict__ x, unsigned short* __restrict__ xbf,
                      unsigned short* __restrict__ xT) {
  __shared__ unsigned short tile[32][33];
  int bid = blockIdx.x;                  // B*(N/32)*(F/32) = 8192
  int fb = bid & 3, nb = (bid >> 2) & 15, b = bid >> 6;
  int tx = threadIdx.x & 31, ty = threadIdx.x >> 5;   // 32 x 8
  const float* xb = x + (size_t)b * N_ * F_;
#pragma unroll
  for (int r = 0; r < 32; r += 8) {
    int n = nb * 32 + ty + r, f = fb * 32 + tx;
    unsigned short h = bf2us((__bf16)xb[(size_t)n * F_ + f]);
    xbf[(size_t)b * N_ * F_ + (size_t)n * F_ + f] = h;
    tile[ty + r][tx] = h;
  }
  __syncthreads();
#pragma unroll
  for (int r = 0; r < 32; r += 8) {
    int f = fb * 32 + ty + r, n = nb * 32 + tx;
    xT[(size_t)b * F_ * N_ + (size_t)f * N_ + n] = tile[tx][ty + r];
  }
}

// M_L -> bf16 K-major:  MLt[n][k] = ML[k][n]
__global__ void k_cvtML(const float* __restrict__ ML, unsigned short* __restrict__ MLt) {
  int idx = blockIdx.x * 256 + threadIdx.x;     // 16384
  int n = idx >> 7, k = idx & 127;
  MLt[idx] = bf2us((__bf16)ML[(size_t)k * F_ + n]);
}

// weight -> bf16 K-major:  wgtT[o][kk] = weight[kk][o]
__global__ void k_cvtW(const float* __restrict__ wgt, unsigned short* __restrict__ wgtT) {
  int idx = blockIdx.x * 256 + threadIdx.x;     // 32768
  int o = idx >> 8, kk = idx & 255;
  wgtT[idx] = bf2us((__bf16)wgt[(size_t)kk * OUT_ + o]);
}

// xw = x @ M_L (bf16 WMMA, f32 accum); sq = rowsum(xw*xw)
__global__ void k_xw(const unsigned short* __restrict__ xbf,
                     const unsigned short* __restrict__ MLt,
                     float* __restrict__ xw, float* __restrict__ sq) {
  int wave = blockIdx.x * (blockDim.x >> 5) + (threadIdx.x >> 5); // 4096 row tiles
  int b = wave >> 5;
  int i0 = (wave & 31) << 4;
  const unsigned short* xb = xbf + (size_t)b * N_ * F_ + (size_t)i0 * F_;
  v8f c[8];
#pragma unroll
  for (int t = 0; t < 8; ++t) c[t] = vzero8();
  for (int k0 = 0; k0 < F_; k0 += 32) {
    v16bf a = a_frag_bf(xb, F_, k0);
#pragma unroll
    for (int ct = 0; ct < 8; ++ct) {
      v16bf bb = b_frag_tr(MLt, F_, k0, ct * 16);
      c[ct] = __builtin_amdgcn_wmma_f32_16x16x32_bf16(false, a, false, bb,
                                                      (short)0, c[ct], false, false);
    }
  }
  int l = threadIdx.x & 31, half = l >> 4, n = l & 15;
  float* xwb = xw + (size_t)b * N_ * F_ + (size_t)i0 * F_;
  float rowsq[8];
#pragma unroll
  for (int v = 0; v < 8; ++v) rowsq[v] = 0.f;
#pragma unroll
  for (int ct = 0; ct < 8; ++ct) {
#pragma unroll
    for (int v = 0; v < 8; ++v) {
      float val = c[ct][v];
      xwb[(size_t)(v + half * 8) * F_ + ct * 16 + n] = val;
      rowsq[v] += val * val;
    }
  }
#pragma unroll
  for (int v = 0; v < 8; ++v) {
    float s = rowsq[v];
    s += __shfl_xor(s, 1); s += __shfl_xor(s, 2);
    s += __shfl_xor(s, 4); s += __shfl_xor(s, 8);
    if (n == 0) sq[(size_t)b * N_ + i0 + v + half * 8] = s;
  }
}

// G = xw @ xw^T via f32 WMMA (cancellation-safe), fused W = exp(-sqrt(d2))*(1-eye)
__global__ void k_gram(const float* __restrict__ xw, const float* __restrict__ sq,
                       float* __restrict__ Wout) {
  int wave = blockIdx.x * (blockDim.x >> 5) + (threadIdx.x >> 5); // 131072 tiles
  int b = wave >> 10;
  int t = wave & 1023;
  int i0 = (t >> 5) << 4;
  int j0 = (t & 31) << 4;
  const float* xwb = xw + (size_t)b * N_ * F_;
  int l = threadIdx.x & 31, half = l >> 4, lm = l & 15;
  const float* ap = xwb + (size_t)(i0 + lm) * F_ + half * 2;
  const float* bp = xwb + (size_t)(j0 + lm) * F_ + half * 2;
  v8f c = vzero8();
  for (int k0 = 0; k0 < F_; k0 += 4) {
    v2f a  = *(const v2f*)(ap + k0);    // A(m,k)
    v2f bb = *(const v2f*)(bp + k0);    // B(k,n) = xw[j][k]
    c = __builtin_amdgcn_wmma_f32_16x16x4_f32(false, a, false, bb,
                                              (short)0, c, false, false);
  }
  float sqj = sq[(size_t)b * N_ + j0 + lm];
  float* Wb = Wout + (size_t)b * N_ * N_;
#pragma unroll
  for (int v = 0; v < 8; ++v) {
    int gi = i0 + v + half * 8;
    int gj = j0 + lm;
    float sqi = sq[(size_t)b * N_ + gi];
    float d2 = fmaxf(sqi + sqj - 2.f * c[v], 0.f);
    float w = (gi == gj) ? 0.f : expf(-sqrtf(d2));
    Wb[(size_t)gi * N_ + gj] = w;
  }
}

// dis = rsqrt(rowsum(W) + TINY)
__global__ void k_dis(const float* __restrict__ Wm, float* __restrict__ dis) {
  int wave = blockIdx.x * (blockDim.x >> 5) + (threadIdx.x >> 5);
  int b = wave >> 9, i = wave & 511;
  const float* row = Wm + (size_t)b * N_ * N_ + (size_t)i * N_;
  int l = threadIdx.x & 31;
  float s = 0.f;
  for (int j = l; j < N_; j += 32) s += row[j];
  s += __shfl_xor(s, 1); s += __shfl_xor(s, 2); s += __shfl_xor(s, 4);
  s += __shfl_xor(s, 8); s += __shfl_xor(s, 16);
  if (l == 0) dis[(size_t)b * N_ + i] = rsqrtf(s + TINY_);
}

// per-batch sum of squares of Lraw = eye - dis_i*W*dis_j
__global__ void k_norm(const float* __restrict__ Wm, const float* __restrict__ dis,
                       float* __restrict__ norm) {
  int b = blockIdx.x >> 7;
  int blk = blockIdx.x & 127;
  const float* Wb = Wm + (size_t)b * N_ * N_;
  const float* db = dis + (size_t)b * N_;
  float acc = 0.f;
  int base = blk * 2048;
#pragma unroll
  for (int e = 0; e < 8; ++e) {
    int idx = base + e * 256 + threadIdx.x;
    int i = idx >> 9, j = idx & 511;
    float t = ((i == j) ? 1.f : 0.f) - db[i] * Wb[idx] * db[j];
    acc += t * t;
  }
  __shared__ float red[8];
  acc += __shfl_xor(acc, 1); acc += __shfl_xor(acc, 2); acc += __shfl_xor(acc, 4);
  acc += __shfl_xor(acc, 8); acc += __shfl_xor(acc, 16);
  if ((threadIdx.x & 31) == 0) red[threadIdx.x >> 5] = acc;
  __syncthreads();
  if (threadIdx.x < 8) {
    float v = red[threadIdx.x];
    v += __shfl_xor(v, 1); v += __shfl_xor(v, 2); v += __shfl_xor(v, 4);
    if (threadIdx.x == 0) atomicAdd(norm + b, v);
  }
}

// L = leakyrelu( (1/max(avg_norm,1)) * (eye - dis_i*W*dis_j) )
__global__ void k_finL(const float* __restrict__ Wm, const float* __restrict__ dis,
                       const float* __restrict__ norm, const float* __restrict__ alpha,
                       float* __restrict__ Lout) {
  int b = blockIdx.x >> 7;
  int blk = blockIdx.x & 127;
  float avg = sqrtf(norm[b]) * (1.f / (float)(N_ * N_));
  float scale = 1.f / fmaxf(avg, 1.f);
  float al = alpha[0];
  const float* Wb = Wm + (size_t)b * N_ * N_;
  const float* db = dis + (size_t)b * N_;
  float* Lb = Lout + (size_t)b * N_ * N_;
  int base = blk * 2048;
#pragma unroll
  for (int e = 0; e < 8; ++e) {
    int idx = base + e * 256 + threadIdx.x;
    int i = idx >> 9, j = idx & 511;
    float t = ((i == j) ? 1.f : 0.f) - db[i] * Wb[idx] * db[j];
    t *= scale;
    Lb[idx] = (t >= 0.f) ? t : al * t;
  }
}

// x1 = (L + laplacian) @ x  (bf16 WMMA; L_all fused, never materialized)
__global__ void k_cheb(const float* __restrict__ L, const float* __restrict__ lap,
                       const unsigned short* __restrict__ xT,
                       unsigned short* __restrict__ x1bf) {
  int wave = blockIdx.x * (blockDim.x >> 5) + (threadIdx.x >> 5);
  int b = wave >> 5;
  int i0 = (wave & 31) << 4;
  const float* Lb = L   + (size_t)b * N_ * N_ + (size_t)i0 * N_;
  const float* Pb = lap + (size_t)b * N_ * N_ + (size_t)i0 * N_;
  const unsigned short* xTb = xT + (size_t)b * F_ * N_;
  v8f c[8];
#pragma unroll
  for (int t = 0; t < 8; ++t) c[t] = vzero8();
  for (int k0 = 0; k0 < N_; k0 += 32) {
    {  // prefetch next K panel of L and lap (global_prefetch_b8)
      int l = threadIdx.x & 31;
      size_t po = (size_t)(l & 15) * N_ + k0 + 32 + (l >> 4) * 8;
      __builtin_prefetch(Lb + po, 0, 0);
      __builtin_prefetch(Pb + po, 0, 0);
    }
    v16bf a = a_frag_sum(Lb, Pb, N_, k0);
#pragma unroll
    for (int ct = 0; ct < 8; ++ct) {
      v16bf bb = b_frag_tr(xTb, N_, k0, ct * 16);
      c[ct] = __builtin_amdgcn_wmma_f32_16x16x32_bf16(false, a, false, bb,
                                                      (short)0, c[ct], false, false);
    }
  }
  int l = threadIdx.x & 31, half = l >> 4, n = l & 15;
  unsigned short* ob = x1bf + (size_t)b * N_ * F_ + (size_t)i0 * F_;
#pragma unroll
  for (int ct = 0; ct < 8; ++ct)
#pragma unroll
    for (int v = 0; v < 8; ++v)
      ob[(size_t)(v + half * 8) * F_ + ct * 16 + n] = bf2us((__bf16)c[ct][v]);
}

// out = relu( xk @ weight + bias ), xk feature index = f*2 + k
// weight^T (bf16, 128x256 K-major, 64KB) staged into LDS via the Tensor Data Mover
__global__ void k_out(const unsigned short* __restrict__ xbf,
                      const unsigned short* __restrict__ x1bf,
                      const unsigned short* __restrict__ wgtT,
                      const float* __restrict__ bias,
                      float* __restrict__ out) {
  __shared__ unsigned short ldsW[OUT_ * F_ * 2];   // 64 KB, layout [o][kk]

  if (threadIdx.x < 32) {   // wave 0 issues one TDM DMA (EXEC ignored by TDM)
    unsigned long long ga = (unsigned long long)(const void*)wgtT;
    v4u g0;
    g0[0] = 1u;                                        // count=1 (valid D#)
    g0[1] = 0u;                                        // lds_addr = 0
    g0[2] = (unsigned)(ga & 0xFFFFFFFFu);              // global_addr[31:0]
    g0[3] = (unsigned)((ga >> 32) & 0x01FFFFFFu)       // global_addr[56:32]
          | (2u << 30);                                // type = 2 (image)
    v8i g1;
    g1[0] = (int)(1u << 16);        // data_size=1 (2B); mask/pad/iterate = 0
    g1[1] = (int)((256u & 0xFFFFu) << 16);   // tensor_dim0 = 256 (bits 79:48)
    g1[2] = (int)((128u & 0xFFFFu) << 16);   // tensor_dim1 = 128 (bits 111:80)
    g1[3] = (int)((256u & 0xFFFFu) << 16);   // tile_dim0   = 256 (bits 127:112)
    g1[4] = 128;                             // tile_dim1   = 128 (bits 143:128)
    g1[5] = 256;                             // tensor_dim0_stride = 256
    g1[6] = 0;
    g1[7] = 0;
    v4i g2; g2[0] = 0; g2[1] = 0; g2[2] = 0; g2[3] = 0;   // 2D: groups 2/3 unused
    v4i g3 = g2;
#if defined(__clang_major__) && (__clang_major__ >= 23)
    v8i g4;
#pragma unroll
    for (int i = 0; i < 8; ++i) g4[i] = 0;
    __builtin_amdgcn_tensor_load_to_lds(g0, g1, g2, g3, g4, 0);
#else
    __builtin_amdgcn_tensor_load_to_lds(g0, g1, g2, g3, 0);
#endif
    __builtin_amdgcn_s_wait_tensorcnt(0);
  }
  __syncthreads();

  int wave = blockIdx.x * (blockDim.x >> 5) + (threadIdx.x >> 5);
  int b = wave >> 5;
  int i0 = (wave & 31) << 4;
  const unsigned short* x0 = xbf  + (size_t)b * N_ * F_ + (size_t)i0 * F_;
  const unsigned short* x1 = x1bf + (size_t)b * N_ * F_ + (size_t)i0 * F_;
  int l = threadIdx.x & 31, half = l >> 4, n = l & 15;
  v8f c[8];
#pragma unroll
  for (int t = 0; t < 8; ++t) c[t] = vzero8();
  for (int k0 = 0; k0 < F_ * 2; k0 += 32) {
    v16bf a = a_frag_inter(x0, x1, k0);
#pragma unroll
    for (int ct = 0; ct < 8; ++ct) {
      const unsigned short* cp = &ldsW[(size_t)(ct * 16 + n) * (F_ * 2) + k0 + half * 8];
      frag_u t;
      t.u[0] = *(const v8us*)(cp);       // ds_load_b128
      t.u[1] = *(const v8us*)(cp + 16);
      c[ct] = __builtin_amdgcn_wmma_f32_16x16x32_bf16(false, a, false, t.b,
                                                      (short)0, c[ct], false, false);
    }
  }
  float* ob = out + (size_t)b * N_ * OUT_ + (size_t)i0 * OUT_;
#pragma unroll
  for (int ct = 0; ct < 8; ++ct) {
    float bi = bias[ct * 16 + n];
#pragma unroll
    for (int v = 0; v < 8; ++v)
      ob[(size_t)(v + half * 8) * OUT_ + ct * 16 + n] = fmaxf(c[ct][v] + bi, 0.f);
  }
}

// ---------- launch ----------
extern "C" void kernel_launch(void* const* d_in, const int* in_sizes, int n_in,
                              void* d_out, int out_size, void* d_ws, size_t ws_size,
                              hipStream_t stream) {
  const float* x     = (const float*)d_in[0];
  const float* lap   = (const float*)d_in[1];
  const float* ML    = (const float*)d_in[2];
  const float* alpha = (const float*)d_in[3];
  const float* wgt   = (const float*)d_in[4];
  const float* bias  = (const float*)d_in[5];

  float* out  = (float*)d_out;                 // [B,N,OUT]
  float* Lout = out + (size_t)B_ * N_ * OUT_;  // [B,N,N]
  float* Wout = Lout + (size_t)B_ * N_ * N_;   // [B,N,N]

  char* w = (char*)d_ws;
  float*          xw   = (float*)w;                       // 33.5 MB (aliased by x1bf)
  unsigned short* x1bf = (unsigned short*)w;              // alias: live after k_gram
  unsigned short* xbf  = (unsigned short*)(w + 33554432); // 16.8 MB  row-major bf16 x
  unsigned short* xT   = (unsigned short*)(w + 50331648); // 16.8 MB  K-major bf16 x
  unsigned short* MLt  = (unsigned short*)(w + 67108864); // 32 KB    K-major bf16 M_L
  unsigned short* wgtT = (unsigned short*)(w + 67141632); // 64 KB    K-major bf16 weight
  float* sq   = (float*)(w + 67207168);                   // 256 KB
  float* dis  = sq + 65536;                               // 256 KB
  float* norm = dis + 65536;                              // 512 B

  k_zero  <<<1,     128, 0, stream>>>(norm);
  k_cvt   <<<8192,  256, 0, stream>>>(x, xbf, xT);
  k_cvtML <<<64,    256, 0, stream>>>(ML, MLt);
  k_cvtW  <<<128,   256, 0, stream>>>(wgt, wgtT);
  k_xw    <<<1024,  128, 0, stream>>>(xbf, MLt, xw, sq);
  k_gram  <<<32768, 128, 0, stream>>>(xw, sq, Wout);
  k_dis   <<<16384, 128, 0, stream>>>(Wout, dis);
  k_norm  <<<16384, 256, 0, stream>>>(Wout, dis, norm);
  k_finL  <<<16384, 256, 0, stream>>>(Wout, dis, norm, alpha, Lout);
  k_cheb  <<<1024,  128, 0, stream>>>(Lout, lap, xT, x1bf);
  k_out   <<<1024,  128, 0, stream>>>(xbf, x1bf, wgtT, bias, out);
}
